// VQVAE_63668595196460
// MI455X (gfx1250) — matrix operand order repaired and checked by reading, not compiled, not run on hardware
//
#include <hip/hip_runtime.h>
#include <hip/hip_bf16.h>
#include <cstdint>
#include <cstddef>

// ---------------------------------------------------------------------------
// Problem constants (from reference)
// ---------------------------------------------------------------------------
#define Bn   256
#define Tn   64
#define Vn   64
#define NIn  512
#define ENHn 1024
#define DNHn 1024
#define NZn  64
#define NDn  8
#define Kn   1000
#define Dn   128
#define H4n  4096   // 4*ENH == 4*DNH

typedef __bf16 bf16_t;
typedef __attribute__((ext_vector_type(16))) __bf16 bf16x16;
typedef __attribute__((ext_vector_type(8)))  __bf16 bf16x8;
typedef __attribute__((ext_vector_type(8)))  float  f32x8;

// ---------------------------------------------------------------------------
// Small utility kernels
// ---------------------------------------------------------------------------
__global__ __launch_bounds__(256) void cast_f32_to_bf16(const float* __restrict__ in,
                                                        bf16_t* __restrict__ out, int n) {
  int i = blockIdx.x * blockDim.x + threadIdx.x;
  if (i < n) out[i] = (bf16_t)in[i];
}

__global__ __launch_bounds__(256) void fill_zero_u32(unsigned* __restrict__ p, int n) {
  int i = blockIdx.x * blockDim.x + threadIdx.x;
  if (i < n) p[i] = 0u;
}

// gather rows of an embedding table, converting to bf16
__global__ __launch_bounds__(128) void gather_embed_bf16(const int* __restrict__ x,
                                                         const float* __restrict__ emb,
                                                         bf16_t* __restrict__ out, int width) {
  int token = blockIdx.x;
  int row = x[token];
  const float* src = emb + (size_t)row * width;
  bf16_t* dst = out + (size_t)token * width;
  for (int w = threadIdx.x; w < width; w += blockDim.x) dst[w] = (bf16_t)src[w];
}

// ---------------------------------------------------------------------------
// WMMA bf16 GEMM:  C[M,N] = sum_s A_s[M,K_s] * W_s[N,K_s]^T  (+ bias[n])
// A, W row-major bf16.  Output f32 (Cf) or bf16 (Cb).
// Wave tile 32x64, block = 4 waves = 64x128.  Requires M%32==0, N%64==0,
// K_s%32==0, lda/ldw multiples of 8/16 (all true for our sizes).
// ---------------------------------------------------------------------------
__device__ __forceinline__ bf16x16 load_a_frag(const bf16_t* p) {
  // lanes 0-15: K 0..7 and 16..23 ; lanes 16-31: K 8..15 and 24..31
  bf16x8 lo = *(const bf16x8*)p;
  bf16x8 hi = *(const bf16x8*)(p + 16);
  bf16x16 r;
#pragma unroll
  for (int i = 0; i < 8; ++i) { r[i] = lo[i]; r[i + 8] = hi[i]; }
  return r;
}

#define WMMA_BF16(A, Bm, C) \
  __builtin_amdgcn_wmma_f32_16x16x32_bf16(false, (A), false, (Bm), (short)0, (C), false, false)

__global__ __launch_bounds__(128) void gemm_bf16_wmma(
    const bf16_t* __restrict__ A0, int lda0, const bf16_t* __restrict__ W0, int ldw0, int K0,
    const bf16_t* __restrict__ A1, int lda1, const bf16_t* __restrict__ W1, int ldw1, int K1,
    const bf16_t* __restrict__ A2, int lda2, const bf16_t* __restrict__ W2, int ldw2, int K2,
    const float* __restrict__ bias, float* __restrict__ Cf, bf16_t* __restrict__ Cb,
    int ldc, int M, int N) {
  const int lane   = threadIdx.x & 31;
  const int wave   = threadIdx.x >> 5;
  const int laneLo = lane & 15;
  const int laneHi = lane >> 4;
  const int m0 = blockIdx.x * 64 + (wave >> 1) * 32;
  const int n0 = blockIdx.y * 128 + (wave & 1) * 64;
  if (m0 >= M || n0 >= N) return;

  f32x8 acc00 = {}, acc01 = {}, acc02 = {}, acc03 = {};
  f32x8 acc10 = {}, acc11 = {}, acc12 = {}, acc13 = {};

  const bf16_t* As[3]  = {A0, A1, A2};
  const bf16_t* Ws[3]  = {W0, W1, W2};
  const int     ldas[3] = {lda0, lda1, lda2};
  const int     ldws[3] = {ldw0, ldw1, ldw2};
  const int     Ks[3]   = {K0, K1, K2};

  for (int s = 0; s < 3; ++s) {
    const int K = Ks[s];
    if (K <= 0) continue;
    const int lda = ldas[s];
    const int ldw = ldws[s];
    const bf16_t* Ab = As[s] + (size_t)(m0 + laneLo) * lda + laneHi * 16;
    const bf16_t* Wb = Ws[s] + (size_t)(n0 + laneLo) * ldw + laneHi * 16;
    for (int k = 0; k < K; k += 32) {
      bf16x16 a0 = *(const bf16x16*)(Ab + k);
      bf16x16 a1 = *(const bf16x16*)(Ab + (size_t)16 * lda + k);
      bf16x16 b0 = *(const bf16x16*)(Wb + k);
      bf16x16 b1 = *(const bf16x16*)(Wb + (size_t)16 * ldw + k);
      bf16x16 b2 = *(const bf16x16*)(Wb + (size_t)32 * ldw + k);
      bf16x16 b3 = *(const bf16x16*)(Wb + (size_t)48 * ldw + k);
      acc00 = WMMA_BF16(a0, b0, acc00);
      acc01 = WMMA_BF16(a0, b1, acc01);
      acc02 = WMMA_BF16(a0, b2, acc02);
      acc03 = WMMA_BF16(a0, b3, acc03);
      acc10 = WMMA_BF16(a1, b0, acc10);
      acc11 = WMMA_BF16(a1, b1, acc11);
      acc12 = WMMA_BF16(a1, b2, acc12);
      acc13 = WMMA_BF16(a1, b3, acc13);
    }
  }

  f32x8 accs[2][4] = {{acc00, acc01, acc02, acc03}, {acc10, acc11, acc12, acc13}};
#pragma unroll
  for (int mi = 0; mi < 2; ++mi) {
#pragma unroll
    for (int ni = 0; ni < 4; ++ni) {
      const int col = n0 + ni * 16 + laneLo;
      const float bv = bias ? bias[col] : 0.0f;
      const size_t base = (size_t)(m0 + mi * 16 + laneHi * 8) * ldc + col;
#pragma unroll
      for (int v = 0; v < 8; ++v) {
        float val = accs[mi][ni][v] + bv;
        if (Cb) Cb[base + (size_t)v * ldc] = (bf16_t)val;
        else    Cf[base + (size_t)v * ldc] = val;
      }
    }
  }
}

// ---------------------------------------------------------------------------
// LSTM pointwise gate update: gates g[B,4H] (i,f,g,o), c[B,H] f32 in/out,
// h written as bf16 (+ optional f32 copy, + optional slot in hs[B,T,H]).
// ---------------------------------------------------------------------------
__global__ __launch_bounds__(256) void lstm_pointwise(
    const float* __restrict__ g, float* __restrict__ c,
    bf16_t* __restrict__ h_bf, float* __restrict__ h_f32,
    bf16_t* __restrict__ hs, unsigned long long hs_off, int H) {
  int idx = blockIdx.x * blockDim.x + threadIdx.x;
  if (idx >= Bn * H) return;
  int b = idx / H;
  int j = idx - b * H;
  const float* gb = g + (size_t)b * 4 * H;
  float gi = gb[j];
  float gf = gb[H + j];
  float gg = gb[2 * H + j];
  float go = gb[3 * H + j];
  float i = 1.0f / (1.0f + __expf(-gi));
  float f = 1.0f / (1.0f + __expf(-gf));
  float t = 2.0f / (1.0f + __expf(-2.0f * gg)) - 1.0f;   // tanh
  float o = 1.0f / (1.0f + __expf(-go));
  float cn = f * c[idx] + i * t;
  c[idx] = cn;
  float tc = 2.0f / (1.0f + __expf(-2.0f * cn)) - 1.0f;  // tanh
  float h = o * tc;
  h_bf[idx] = (bf16_t)h;
  if (h_f32) h_f32[idx] = h;
  if (hs) hs[hs_off + (size_t)b * ((size_t)Tn * H) + j] = (bf16_t)h;
}

// ---------------------------------------------------------------------------
// VQ: per (b, nd) find nearest of 1000 codebook rows (D=128), write q (bf16)
// and vq_loss = (1+BETA) * mean((q-flat)^2).
// ---------------------------------------------------------------------------
__global__ __launch_bounds__(128) void vq_kernel(const float* __restrict__ hb,
                                                 const float* __restrict__ cb,
                                                 bf16_t* __restrict__ q_bf,
                                                 float* __restrict__ vq_out) {
  const int bid = blockIdx.x;       // B*ND blocks
  const int b = bid >> 3;
  const int nd = bid & 7;
  __shared__ float fvec[Dn];
  __shared__ float wbest[4];
  __shared__ int   wbestk[4];
  __shared__ int   kstar_sh;
  const int tid = threadIdx.x;
  fvec[tid] = hb[(size_t)b * (NDn * Dn) + nd * Dn + tid];
  __syncthreads();
  const int lane = tid & 31;
  const int wave = tid >> 5;
  const float* cbn = cb + (size_t)nd * Kn * Dn;
  float best = 3.4e38f;
  int bestk = 0;
  for (int k = wave; k < Kn; k += 4) {
    const float* ck = cbn + (size_t)k * Dn;
    float d = 0.0f;
#pragma unroll
    for (int u = 0; u < 4; ++u) {
      float diff = fvec[lane * 4 + u] - ck[lane * 4 + u];
      d += diff * diff;
    }
#pragma unroll
    for (int off = 16; off > 0; off >>= 1) d += __shfl_xor(d, off, 32);
    if (d < best) { best = d; bestk = k; }
  }
  if (lane == 0) { wbest[wave] = best; wbestk[wave] = bestk; }
  __syncthreads();
  if (tid == 0) {
    float bb = wbest[0]; int kk = wbestk[0];
    for (int w = 1; w < 4; ++w)
      if (wbest[w] < bb) { bb = wbest[w]; kk = wbestk[w]; }
    kstar_sh = kk;
    vq_out[b * NDn + nd] = 1.25f * bb / (float)Dn;   // (1+BETA)*mean
  }
  __syncthreads();
  const int kstar = kstar_sh;
  q_bf[(size_t)b * (NDn * Dn) + nd * Dn + tid] = (bf16_t)cbn[(size_t)kstar * Dn + tid];
}

// ---------------------------------------------------------------------------
// KL: muv[B,128] -> kl[b] = 0.5*sum_j mu^2 + exp(lv) - lv - 1 ; also mu->bf16
// ---------------------------------------------------------------------------
__global__ __launch_bounds__(64) void kl_kernel(const float* __restrict__ muv,
                                                bf16_t* __restrict__ mu_bf,
                                                float* __restrict__ kl_out) {
  const int b = blockIdx.x;
  const int j = threadIdx.x;   // 64 threads
  float m  = muv[(size_t)b * 128 + j];
  float lv = muv[(size_t)b * 128 + 64 + j];
  mu_bf[(size_t)b * 64 + j] = (bf16_t)m;
  float term = m * m + __expf(lv) - lv - 1.0f;
#pragma unroll
  for (int off = 16; off > 0; off >>= 1) term += __shfl_xor(term, off, 32);
  __shared__ float s[2];
  if ((j & 31) == 0) s[j >> 5] = term;
  __syncthreads();
  if (j == 0) kl_out[b] = 0.5f * (s[0] + s[1]);
}

// ---------------------------------------------------------------------------
// Host orchestration
// ---------------------------------------------------------------------------
extern "C" void kernel_launch(void* const* d_in, const int* in_sizes, int n_in,
                              void* d_out, int out_size, void* d_ws, size_t ws_size,
                              hipStream_t stream) {
  (void)in_sizes; (void)n_in; (void)out_size; (void)ws_size;

  const int*   x         = (const int*)  d_in[0];
  const float* enc_emb   = (const float*)d_in[1];
  const float* enc_Wih_f = (const float*)d_in[2];
  const float* enc_Whh_f = (const float*)d_in[3];
  const float* enc_b_f   = (const float*)d_in[4];
  const float* enc_Wih_b = (const float*)d_in[5];
  const float* enc_Whh_b = (const float*)d_in[6];
  const float* enc_b_b   = (const float*)d_in[7];
  const float* enc_lin   = (const float*)d_in[8];
  const float* z2d_W     = (const float*)d_in[9];
  const float* z2d_b     = (const float*)d_in[10];
  const float* codebooks = (const float*)d_in[11];
  const float* dec_emb   = (const float*)d_in[12];
  const float* dec_Wih   = (const float*)d_in[13];
  const float* dec_Whh   = (const float*)d_in[14];
  const float* dec_b     = (const float*)d_in[15];
  const float* pred_W    = (const float*)d_in[16];

  float* out_logits = (float*)d_out;                           // [B,T,V]
  float* out_vq     = out_logits + (size_t)Bn * Tn * Vn;       // [B,ND]
  float* out_kl     = out_vq + (size_t)Bn * NDn;               // [B]

  // ---- workspace bump allocator ----
  char* wp = (char*)d_ws;
  auto alloc = [&](size_t bytes) -> void* {
    void* r = (void*)wp;
    wp += (bytes + 255) & ~(size_t)255;
    return r;
  };
  const size_t NT = (size_t)Bn * Tn;   // 16384 tokens
  bf16_t* we_bf    = (bf16_t*)alloc(NT * NIn * 2);
  bf16_t* de_bf    = (bf16_t*)alloc(NT * NIn * 2);
  bf16_t* Wihf_bf  = (bf16_t*)alloc((size_t)H4n * NIn * 2);
  bf16_t* Whhf_bf  = (bf16_t*)alloc((size_t)H4n * ENHn * 2);
  bf16_t* Wihb_bf  = (bf16_t*)alloc((size_t)H4n * NIn * 2);
  bf16_t* Whhb_bf  = (bf16_t*)alloc((size_t)H4n * ENHn * 2);
  bf16_t* dWih_bf  = (bf16_t*)alloc((size_t)H4n * (NIn + ENHn) * 2);
  bf16_t* dWhh_bf  = (bf16_t*)alloc((size_t)H4n * DNHn * 2);
  bf16_t* elin_bf  = (bf16_t*)alloc((size_t)2 * NZn * ENHn * 2);
  bf16_t* z2d_bf   = (bf16_t*)alloc((size_t)DNHn * NZn * 2);
  bf16_t* pred_bf  = (bf16_t*)alloc((size_t)Vn * DNHn * 2);
  float*  g_ws     = (float*) alloc((size_t)Bn * H4n * 4);
  bf16_t* hf_bf    = (bf16_t*)alloc((size_t)Bn * ENHn * 2);
  float*  cf_ws    = (float*) alloc((size_t)Bn * ENHn * 4);
  bf16_t* hb_bf    = (bf16_t*)alloc((size_t)Bn * ENHn * 2);
  float*  cb_ws    = (float*) alloc((size_t)Bn * ENHn * 4);
  float*  hb_f32   = (float*) alloc((size_t)Bn * ENHn * 4);
  bf16_t* hd_bf    = (bf16_t*)alloc((size_t)Bn * DNHn * 2);
  float*  cd_ws    = (float*) alloc((size_t)Bn * DNHn * 4);
  bf16_t* hs_bf    = (bf16_t*)alloc(NT * DNHn * 2);
  float*  muv_ws   = (float*) alloc((size_t)Bn * 128 * 4);
  bf16_t* mu_bf    = (bf16_t*)alloc((size_t)Bn * NZn * 2);
  bf16_t* q_bf     = (bf16_t*)alloc((size_t)Bn * ENHn * 2);

  // ---- launch helpers ----
  auto cast = [&](const float* in, bf16_t* outp, size_t n) {
    cast_f32_to_bf16<<<dim3((unsigned)((n + 255) / 256)), dim3(256), 0, stream>>>(in, outp, (int)n);
  };
  auto zero = [&](void* p, size_t n_u32) {
    fill_zero_u32<<<dim3((unsigned)((n_u32 + 255) / 256)), dim3(256), 0, stream>>>((unsigned*)p, (int)n_u32);
  };
  auto gemm = [&](const bf16_t* A0, int lda0, const bf16_t* W0, int ldw0, int K0,
                  const bf16_t* A1, int lda1, const bf16_t* W1, int ldw1, int K1,
                  const bf16_t* A2, int lda2, const bf16_t* W2, int ldw2, int K2,
                  const float* bias, float* Cf, bf16_t* Cb, int ldc, int M, int N) {
    dim3 grid((unsigned)((M + 63) / 64), (unsigned)((N + 127) / 128));
    gemm_bf16_wmma<<<grid, dim3(128), 0, stream>>>(A0, lda0, W0, ldw0, K0,
                                                   A1, lda1, W1, ldw1, K1,
                                                   A2, lda2, W2, ldw2, K2,
                                                   bias, Cf, Cb, ldc, M, N);
  };
  auto pointwise = [&](float* c, bf16_t* h_bf, float* h_f32, bf16_t* hs,
                       unsigned long long hs_off) {
    lstm_pointwise<<<dim3((Bn * ENHn) / 256), dim3(256), 0, stream>>>(
        g_ws, c, h_bf, h_f32, hs, hs_off, ENHn);
  };

  // ---- 1. weight casts + embedding gathers ----
  cast(enc_Wih_f, Wihf_bf, (size_t)H4n * NIn);
  cast(enc_Whh_f, Whhf_bf, (size_t)H4n * ENHn);
  cast(enc_Wih_b, Wihb_bf, (size_t)H4n * NIn);
  cast(enc_Whh_b, Whhb_bf, (size_t)H4n * ENHn);
  cast(dec_Wih,   dWih_bf, (size_t)H4n * (NIn + ENHn));
  cast(dec_Whh,   dWhh_bf, (size_t)H4n * DNHn);
  cast(enc_lin,   elin_bf, (size_t)2 * NZn * ENHn);
  cast(z2d_W,     z2d_bf,  (size_t)DNHn * NZn);
  cast(pred_W,    pred_bf, (size_t)Vn * DNHn);
  gather_embed_bf16<<<dim3((unsigned)NT), dim3(128), 0, stream>>>(x, enc_emb, we_bf, NIn);
  gather_embed_bf16<<<dim3((unsigned)NT), dim3(128), 0, stream>>>(x, dec_emb, de_bf, NIn);

  // ---- 2. encoder forward LSTM ----
  zero(hf_bf, (size_t)Bn * ENHn / 2);
  zero(cf_ws, (size_t)Bn * ENHn);
  for (int t = 0; t < Tn; ++t) {
    gemm(we_bf + (size_t)t * NIn, Tn * NIn, Wihf_bf, NIn, NIn,
         hf_bf, ENHn, Whhf_bf, ENHn, ENHn,
         nullptr, 0, nullptr, 0, 0,
         enc_b_f, g_ws, nullptr, H4n, Bn, H4n);
    pointwise(cf_ws, hf_bf, nullptr, nullptr, 0ULL);
  }

  // ---- 3. encoder backward LSTM (reversed time) ----
  zero(hb_bf, (size_t)Bn * ENHn / 2);
  zero(cb_ws, (size_t)Bn * ENHn);
  for (int t = 0; t < Tn; ++t) {
    const int tr = Tn - 1 - t;
    gemm(we_bf + (size_t)tr * NIn, Tn * NIn, Wihb_bf, NIn, NIn,
         hb_bf, ENHn, Whhb_bf, ENHn, ENHn,
         nullptr, 0, nullptr, 0, 0,
         enc_b_b, g_ws, nullptr, H4n, Bn, H4n);
    pointwise(cb_ws, hb_bf, hb_f32, nullptr, 0ULL);
  }

  // ---- 4. latent head: muv = hf @ enc_lin^T ; kl + mu->bf16 ----
  gemm(hf_bf, ENHn, elin_bf, ENHn, ENHn,
       nullptr, 0, nullptr, 0, 0, nullptr, 0, nullptr, 0, 0,
       nullptr, muv_ws, nullptr, 128, Bn, 128);
  kl_kernel<<<dim3(Bn), dim3(64), 0, stream>>>(muv_ws, mu_bf, out_kl);

  // ---- 5. decoder h0 = mu @ z2d_W^T + z2d_b (bf16 out); c0 = 0 ----
  gemm(mu_bf, NZn, z2d_bf, NZn, NZn,
       nullptr, 0, nullptr, 0, 0, nullptr, 0, nullptr, 0, 0,
       z2d_b, nullptr, hd_bf, DNHn, Bn, DNHn);
  zero(cd_ws, (size_t)Bn * DNHn);

  // ---- 6. VQ: nearest codebook rows from hb, write q (bf16) + vq_loss ----
  vq_kernel<<<dim3(Bn * NDn), dim3(Dn), 0, stream>>>(hb_f32, codebooks, q_bf, out_vq);

  // ---- 7. decoder LSTM: dec_in = [de_t | q] ; g = dec_in@Wih^T + h@Whh^T + b
  for (int t = 0; t < Tn; ++t) {
    gemm(de_bf + (size_t)t * NIn, Tn * NIn, dWih_bf,       NIn + ENHn, NIn,
         q_bf,  ENHn,             dWih_bf + NIn, NIn + ENHn, ENHn,
         hd_bf, DNHn,             dWhh_bf,       DNHn,       DNHn,
         dec_b, g_ws, nullptr, H4n, Bn, H4n);
    pointwise(cd_ws, hd_bf, nullptr, hs_bf, (unsigned long long)t * DNHn);
  }

  // ---- 8. logits = hs @ pred_W^T ----
  gemm(hs_bf, DNHn, pred_bf, DNHn, DNHn,
       nullptr, 0, nullptr, 0, 0, nullptr, 0, nullptr, 0, 0,
       nullptr, out_logits, nullptr, Vn, (int)NT, Vn);
}
